// SlidingWindowMultiheadAttention_45397804319138
// MI455X (gfx1250) — matrix operand order repaired and checked by reading, not compiled, not run on hardware
//
#include <hip/hip_runtime.h>

// Sliding-window MHA on gfx1250:
//   f32->bf16 pre-convert -> QKV GEMM (WMMA bf16, async-LDS double-buffered)
//   -> windowed attention (VALU, bf16 ctx out) -> out GEMM (WMMA bf16)

#define HIDDEN 1024
#define NHEADS 16
#define HDIM   64
#define WIN    33
#define BATCH  2

typedef __bf16 v16bf __attribute__((ext_vector_type(16)));
typedef float  v8f   __attribute__((ext_vector_type(8)));

__device__ __forceinline__ unsigned short f32_bf16(float f) {
    unsigned u = __builtin_bit_cast(unsigned, f);
    u += 0x7fffu + ((u >> 16) & 1u);          // round-to-nearest-even
    return (unsigned short)(u >> 16);
}

union FragAB { v16bf v; uint4 q[2]; };

// CDNA5 async global->LDS copy, 16B per lane, tracked by ASYNCcnt.
// ISA 10.2: generic LDS address low 32 bits == LDS byte offset.
__device__ __forceinline__ void async_ld16(unsigned lds_addr, const void* sbase, unsigned goff) {
    asm volatile("global_load_async_to_lds_b128 %0, %1, %2 offset:0"
                 :: "v"(lds_addr), "v"(goff), "s"(sbase) : "memory");
}
__device__ __forceinline__ void wait_async0() {
    asm volatile("s_wait_asynccnt 0x0" ::: "memory");
}

#define LDSTRIDE 40   // halves per LDS row: 32 data + 8 pad (80B = 20 dwords, 16B aligned)

__device__ __forceinline__ void compute_tile(const unsigned short* __restrict__ lA,
                                             const unsigned short* __restrict__ lB,
                                             int wm, int wn, int l16, int hi,
                                             v8f acc[4][2])
{
    FragAB af[4], bfr[2];
#pragma unroll
    for (int mt = 0; mt < 4; ++mt) {
        int r = wm * 64 + mt * 16 + l16;
        // A 16x32 bf16 frag: e0-7 = K[hi*8..], e8-15 = K[16+hi*8..]
        af[mt].q[0] = *(const uint4*)&lA[r * LDSTRIDE + hi * 8];
        af[mt].q[1] = *(const uint4*)&lA[r * LDSTRIDE + 16 + hi * 8];
    }
#pragma unroll
    for (int nt = 0; nt < 2; ++nt) {
        int r = wn * 32 + nt * 16 + l16;
        // B 32x16 bf16 frag: e0-7 = K[hi*8..], e8-15 = K[16+hi*8..]
        bfr[nt].q[0] = *(const uint4*)&lB[r * LDSTRIDE + hi * 8];
        bfr[nt].q[1] = *(const uint4*)&lB[r * LDSTRIDE + 16 + hi * 8];
    }
#pragma unroll
    for (int mt = 0; mt < 4; ++mt)
#pragma unroll
        for (int nt = 0; nt < 2; ++nt)
            acc[mt][nt] = __builtin_amdgcn_wmma_f32_16x16x32_bf16(
                false, af[mt].v, false, bfr[nt].v,
                (short)0, acc[mt][nt], false, false);
}

// C[M,N] = A[M,K] @ W[N,K]^T + bias[N]; A,W bf16 row-major, C f32.
__global__ __launch_bounds__(256)
void gemm_bf16_wmma(const unsigned short* __restrict__ A,
                    const unsigned short* __restrict__ W,
                    const float* __restrict__ bias, float* __restrict__ C,
                    int M, int N, int K)
{
    __shared__ __align__(16) unsigned short ldsA[2][128 * LDSTRIDE];
    __shared__ __align__(16) unsigned short ldsB[2][128 * LDSTRIDE];

    const int tid  = threadIdx.x;
    const int lane = tid & 31;
    const int wid  = tid >> 5;
    const int wm   = wid & 1;     // 2 wave-rows of 64
    const int wn   = wid >> 1;    // 4 wave-cols of 32
    const int hi   = lane >> 4;
    const int l16  = lane & 15;

    const int bm = blockIdx.y * 128;
    const int bn = blockIdx.x * 128;

    // Each thread copies 2 x 16B per matrix per tile (128x32 bf16 = 512 units).
    unsigned goffA[2], goffW[2];
    unsigned ldsAaddr[2][2], ldsBaddr[2][2];   // [buf][unit]
#pragma unroll
    for (int u2 = 0; u2 < 2; ++u2) {
        int u   = tid + u2 * 256;
        int row = u >> 2;
        int c16 = u & 3;
        goffA[u2] = (unsigned)((bm + row) * K + c16 * 8) * 2u;   // bytes
        goffW[u2] = (unsigned)((bn + row) * K + c16 * 8) * 2u;
#pragma unroll
        for (int b = 0; b < 2; ++b) {
            ldsAaddr[b][u2] = (unsigned)(size_t)&ldsA[b][row * LDSTRIDE + c16 * 8];
            ldsBaddr[b][u2] = (unsigned)(size_t)&ldsB[b][row * LDSTRIDE + c16 * 8];
        }
    }

    v8f acc[4][2];
    v8f vzero = {0.f,0.f,0.f,0.f,0.f,0.f,0.f,0.f};
#pragma unroll
    for (int i = 0; i < 4; ++i)
#pragma unroll
        for (int j = 0; j < 2; ++j) acc[i][j] = vzero;

#define STAGE(buf, kk)                                                    \
    do {                                                                  \
        unsigned kb = (unsigned)(kk) * 2u;                                \
        _Pragma("unroll")                                                 \
        for (int u2 = 0; u2 < 2; ++u2) {                                  \
            async_ld16(ldsAaddr[buf][u2], A, goffA[u2] + kb);             \
            async_ld16(ldsBaddr[buf][u2], W, goffW[u2] + kb);             \
        }                                                                 \
    } while (0)

    // prologue
    STAGE(0, 0);
    wait_async0();
    __syncthreads();

    for (int k0 = 0; k0 < K; k0 += 64) {       // K % 64 == 0
        STAGE(1, k0 + 32);
        compute_tile(ldsA[0], ldsB[0], wm, wn, l16, hi, acc);
        wait_async0();
        __syncthreads();
        if (k0 + 64 < K) STAGE(0, k0 + 64);
        compute_tile(ldsA[1], ldsB[1], wm, wn, l16, hi, acc);
        wait_async0();
        __syncthreads();
    }
#undef STAGE

    // epilogue: C/D layout -> lane l16 = col, vgpr r = row (r + 8*hi)
#pragma unroll
    for (int mt = 0; mt < 4; ++mt) {
#pragma unroll
        for (int nt = 0; nt < 2; ++nt) {
            int col = bn + wn * 32 + nt * 16 + l16;
            float bv = bias ? bias[col] : 0.0f;
#pragma unroll
            for (int r = 0; r < 8; ++r) {
                int row = bm + wm * 64 + mt * 16 + hi * 8 + r;
                C[(size_t)row * N + col] = acc[mt][nt][r] + bv;
            }
        }
    }
}

// f32 -> bf16 pack, 4 elements per thread
__global__ __launch_bounds__(256)
void cvt_bf16(const float* __restrict__ in, unsigned short* __restrict__ out, int n4)
{
    int i = blockIdx.x * 256 + threadIdx.x;
    if (i >= n4) return;
    float4 v = ((const float4*)in)[i];
    uint2 p;
    p.x = (unsigned)f32_bf16(v.x) | ((unsigned)f32_bf16(v.y) << 16);
    p.y = (unsigned)f32_bf16(v.z) | ((unsigned)f32_bf16(v.w) << 16);
    ((uint2*)out)[i] = p;
}

// One thread per query; K/V window in LDS. Zero-padded edges reproduce the
// reference exactly (padded score = 0 participates in softmax, padded V = 0).
// Writes ctx in bf16 (A-operand of the output GEMM).
#define QBLK 64
#define KROWS (QBLK + 32)    // 96
#define ASTRIDE 68           // padded float stride (272B rows, 16B aligned)

__global__ __launch_bounds__(64)
void swin_attn(const float* __restrict__ qkv, unsigned short* __restrict__ ctxb, int S)
{
    __shared__ __align__(16) float kbuf[KROWS * ASTRIDE];
    __shared__ __align__(16) float vbuf[KROWS * ASTRIDE];

    const int bh = blockIdx.x;
    const int b  = bh / NHEADS;
    const int h  = bh % NHEADS;
    const int s0 = blockIdx.y * QBLK;
    const int tid = threadIdx.x;

    for (int i = tid; i < KROWS * (HDIM / 4); i += QBLK) {
        int row = i >> 4;
        int c   = (i & 15) << 2;
        int s   = s0 - 16 + row;
        float4 kv = make_float4(0.f, 0.f, 0.f, 0.f);
        float4 vv = kv;
        if (s >= 0 && s < S) {
            const float* base = qkv + ((size_t)(b * S + s)) * (3 * HIDDEN) + h * (3 * HDIM);
            kv = *(const float4*)(base + HDIM + c);
            vv = *(const float4*)(base + 2 * HDIM + c);
        }
        *(float4*)&kbuf[row * ASTRIDE + c] = kv;
        *(float4*)&vbuf[row * ASTRIDE + c] = vv;
    }
    __syncthreads();

    const int s = s0 + tid;
    const float* qb = qkv + ((size_t)(b * S + s)) * (3 * HIDDEN) + h * (3 * HDIM);
    float4 q[16];
#pragma unroll
    for (int d = 0; d < 16; ++d) q[d] = *(const float4*)(qb + d * 4);

    float sc[WIN];
#pragma unroll 4
    for (int w = 0; w < WIN; ++w) {
        const float* kr = &kbuf[(tid + w) * ASTRIDE];
        float a = 0.f;
#pragma unroll
        for (int d = 0; d < 16; ++d) {
            float4 kk = *(const float4*)(kr + d * 4);
            a += q[d].x * kk.x + q[d].y * kk.y + q[d].z * kk.z + q[d].w * kk.w;
        }
        sc[w] = a * 0.125f;   // 1/sqrt(64)
    }

    float m = sc[0];
#pragma unroll
    for (int w = 1; w < WIN; ++w) m = fmaxf(m, sc[w]);
    float sum = 0.f;
#pragma unroll
    for (int w = 0; w < WIN; ++w) { sc[w] = __expf(sc[w] - m); sum += sc[w]; }
    float inv = 1.0f / sum;
#pragma unroll
    for (int w = 0; w < WIN; ++w) sc[w] *= inv;

    unsigned short* cb = ctxb + ((size_t)(b * S + s)) * HIDDEN + h * HDIM;
#pragma unroll
    for (int d = 0; d < 16; ++d) {
        float ax = 0.f, ay = 0.f, az = 0.f, aw = 0.f;
#pragma unroll 4
        for (int w = 0; w < WIN; ++w) {
            float4 vv = *(const float4*)&vbuf[(tid + w) * ASTRIDE + d * 4];
            float at = sc[w];
            ax += at * vv.x; ay += at * vv.y; az += at * vv.z; aw += at * vv.w;
        }
        uint2 p;
        p.x = (unsigned)f32_bf16(ax) | ((unsigned)f32_bf16(ay) << 16);
        p.y = (unsigned)f32_bf16(az) | ((unsigned)f32_bf16(aw) << 16);
        *(uint2*)(cb + d * 4) = p;
    }
}

extern "C" void kernel_launch(void* const* d_in, const int* in_sizes, int n_in,
                              void* d_out, int out_size, void* d_ws, size_t ws_size,
                              hipStream_t stream)
{
    const float* x     = (const float*)d_in[0];
    const float* w_qkv = (const float*)d_in[1];
    const float* b_qkv = (const float*)d_in[2];
    const float* w_out = (const float*)d_in[3];
    const float* b_out = (const float*)d_in[4];
    float* out = (float*)d_out;

    const int M = in_sizes[0] / HIDDEN;   // B*S = 4096
    const int S = M / BATCH;              // 2048

    char* ws = (char*)d_ws;
    unsigned short* xb    = (unsigned short*)(ws);                          // M x 1024 bf16
    unsigned short* wqkvb = (unsigned short*)(ws + (size_t)8  * 1024*1024); // 3072 x 1024 bf16
    unsigned short* woutb = (unsigned short*)(ws + (size_t)14 * 1024*1024); // 1024 x 1024 bf16
    float*          qkv   = (float*)        (ws + (size_t)16 * 1024*1024);  // M x 3072 f32
    unsigned short* ctxb  = (unsigned short*)(ws + (size_t)64 * 1024*1024); // M x 1024 bf16

    const int nx  = M * HIDDEN / 4;
    const int nwq = 3 * HIDDEN * HIDDEN / 4;
    const int nwo = HIDDEN * HIDDEN / 4;
    cvt_bf16<<<dim3((nx  + 255) / 256), dim3(256), 0, stream>>>(x,     xb,    nx);
    cvt_bf16<<<dim3((nwq + 255) / 256), dim3(256), 0, stream>>>(w_qkv, wqkvb, nwq);
    cvt_bf16<<<dim3((nwo + 255) / 256), dim3(256), 0, stream>>>(w_out, woutb, nwo);

    // qkv = x @ w_qkv^T + b_qkv
    gemm_bf16_wmma<<<dim3((3 * HIDDEN) / 128, M / 128), dim3(256), 0, stream>>>(
        xb, wqkvb, b_qkv, qkv, M, 3 * HIDDEN, HIDDEN);
    // sliding-window attention per (b, h)
    swin_attn<<<dim3(BATCH * NHEADS, S / QBLK), dim3(QBLK), 0, stream>>>(qkv, ctxb, S);
    // out = ctx @ w_out^T + b_out  (final output stays f32)
    gemm_bf16_wmma<<<dim3(HIDDEN / 128, M / 128), dim3(256), 0, stream>>>(
        ctxb, woutb, b_out, out, M, HIDDEN, HIDDEN);
}
